// SelfAttention2d_65008624992546
// MI455X (gfx1250) — matrix operand order repaired and checked by reading, not compiled, not run on hardware
//
#include <hip/hip_runtime.h>
#include <math.h>

typedef _Float16 half_t;
typedef __attribute__((ext_vector_type(16))) _Float16 v16h;
typedef __attribute__((ext_vector_type(8)))  _Float16 v8h;
typedef __attribute__((ext_vector_type(8)))  float    v8f;
typedef __attribute__((ext_vector_type(4)))  int      v4i;

#define B_SZ   8
#define C_SZ   512
#define NH_SZ  8
#define HD_SZ  64
#define N_SZ   1024
#define G_SZ   8
#define CPG_SZ 64

// ---- WMMA wrappers (reuse_a hint legal when next instruction is identical) ----
__device__ __forceinline__ v8f wmma_f16(v16h a, v16h b, v8f c) {
  return __builtin_amdgcn_wmma_f32_16x16x32_f16(false, a, false, b, (short)0, c, false, false);
}
__device__ __forceinline__ v8f wmma_f16_ra(v16h a, v16h b, v8f c) {
  return __builtin_amdgcn_wmma_f32_16x16x32_f16(false, a, false, b, (short)0, c, true, false);
}

__device__ __forceinline__ v8f zero_v8f() {
  v8f z;
#pragma unroll
  for (int i = 0; i < 8; ++i) z[i] = 0.0f;
  return z;
}

// ---- async global->LDS staging (gfx1250) ----
#if __has_builtin(__builtin_amdgcn_global_load_async_to_lds_b128)
__device__ __forceinline__ void async_copy16(const half_t* g, half_t* l) {
  __builtin_amdgcn_global_load_async_to_lds_b128(
      (__attribute__((address_space(1))) v4i*)g,
      (__attribute__((address_space(3))) v4i*)l, 0, 0);
}
#else
__device__ __forceinline__ void async_copy16(const half_t* g, half_t* l) {
  *(v8h*)l = *(const v8h*)g;   // plain load + ds_store fallback
}
#endif

#if __has_builtin(__builtin_amdgcn_s_wait_asynccnt)
#define WAIT_ASYNC(n) __builtin_amdgcn_s_wait_asynccnt(n)
#else
#define WAIT_ASYNC(n) asm volatile("s_wait_asynccnt %0" ::"i"(n) : "memory")
#endif

// A-fragment (16x32 f16) from row-major f16 matrix (row = M, ld in elements).
// VGPR j<4: K=2j..2j+1 (+8 for lanes 16-31); j>=4: K=16+2(j-4).. (+8).
__device__ __forceinline__ v16h load_a_f16(const half_t* __restrict__ p, int ld, int lane) {
  const int m = lane & 15, hh = lane >> 4;
  const half_t* rp = p + (size_t)m * ld;
  v16h a;
#pragma unroll
  for (int j = 0; j < 8; ++j) {
    const int kb = 2 * (j & 3) + 16 * (j >> 2) + 8 * hh;
    a[2 * j]     = rp[kb];
    a[2 * j + 1] = rp[kb + 1];
  }
  return a;
}

// A-fragment from row-major f32 matrix (weights), converted to f16 on load.
__device__ __forceinline__ v16h load_a_f32(const float* __restrict__ p, int ld, int lane) {
  const int m = lane & 15, hh = lane >> 4;
  const float* rp = p + (size_t)m * ld;
  v16h a;
#pragma unroll
  for (int j = 0; j < 8; ++j) {
    const int kb = 2 * (j & 3) + 16 * (j >> 2) + 8 * hh;
    a[2 * j]     = (half_t)rp[kb];
    a[2 * j + 1] = (half_t)rp[kb + 1];
  }
  return a;
}

// B-fragment (32x16 f16) from "K-contiguous" storage X[n][k] (n = B column).
// lanes 0-15 hold K=0..7,16..23, lanes 16-31 hold K=8..15,24..31; two 16-byte runs per lane.
__device__ __forceinline__ v16h load_b_kc(const half_t* __restrict__ p, int ld, int lane) {
  const int n = lane & 15, koff = (lane >> 4) * 8;
  const half_t* rp = p + (size_t)n * ld + koff;
  v16h b;
#pragma unroll
  for (int j = 0; j < 16; ++j) b[j] = rp[(j & 7) + 16 * (j >> 3)];
  return b;
}

// 64x128-tile GEMM inner loop over K=C_SZ, unroll-by-2 double buffer (no
// register rotation -> no v_mov churn). A = f32 weights row-major; B = f16
// K-contiguous [n][k]. acc = 4 C-frags (16 x 64 output per wave).
__device__ __forceinline__ void gemm_k_loop(const float* __restrict__ Arow,
                                            const half_t* __restrict__ Bbase,
                                            int lane, v8f acc[4]) {
  v16h aA  = load_a_f32(Arow, C_SZ, lane);
  v16h bA0 = load_b_kc(Bbase + (size_t)0  * C_SZ, C_SZ, lane);
  v16h bA1 = load_b_kc(Bbase + (size_t)16 * C_SZ, C_SZ, lane);
  v16h bA2 = load_b_kc(Bbase + (size_t)32 * C_SZ, C_SZ, lane);
  v16h bA3 = load_b_kc(Bbase + (size_t)48 * C_SZ, C_SZ, lane);
  for (int k = 0; k < C_SZ; k += 64) {
    // issue set-B loads (k+32, always in range) before consuming set A
    __builtin_prefetch(Arow + k + 64, 0, 1);
    const v16h aB  = load_a_f32(Arow + k + 32, C_SZ, lane);
    const v16h bB0 = load_b_kc(Bbase + (size_t)0  * C_SZ + k + 32, C_SZ, lane);
    const v16h bB1 = load_b_kc(Bbase + (size_t)16 * C_SZ + k + 32, C_SZ, lane);
    const v16h bB2 = load_b_kc(Bbase + (size_t)32 * C_SZ + k + 32, C_SZ, lane);
    const v16h bB3 = load_b_kc(Bbase + (size_t)48 * C_SZ + k + 32, C_SZ, lane);
    acc[0] = wmma_f16_ra(aA, bA0, acc[0]);
    acc[1] = wmma_f16_ra(aA, bA1, acc[1]);
    acc[2] = wmma_f16_ra(aA, bA2, acc[2]);
    acc[3] = wmma_f16   (aA, bA3, acc[3]);
    if (k + 64 < C_SZ) {  // issue set-A loads for next iteration
      aA  = load_a_f32(Arow + k + 64, C_SZ, lane);
      bA0 = load_b_kc(Bbase + (size_t)0  * C_SZ + k + 64, C_SZ, lane);
      bA1 = load_b_kc(Bbase + (size_t)16 * C_SZ + k + 64, C_SZ, lane);
      bA2 = load_b_kc(Bbase + (size_t)32 * C_SZ + k + 64, C_SZ, lane);
      bA3 = load_b_kc(Bbase + (size_t)48 * C_SZ + k + 64, C_SZ, lane);
    }
    acc[0] = wmma_f16_ra(aB, bB0, acc[0]);
    acc[1] = wmma_f16_ra(aB, bB1, acc[1]);
    acc[2] = wmma_f16_ra(aB, bB2, acc[2]);
    acc[3] = wmma_f16   (aB, bB3, acc[3]);
  }
}

// ---------------------------------------------------------------------------
// Kernel 1: GroupNorm -> xnT[b][n][c] (f16, K-contiguous for the QKV GEMMs)
// ---------------------------------------------------------------------------
__global__ void groupnorm_k(const float* __restrict__ x, const float* __restrict__ w,
                            const float* __restrict__ bias, half_t* __restrict__ xnT) {
  const int b = blockIdx.x / G_SZ;
  const int g = blockIdx.x % G_SZ;
  const float* xg = x + ((size_t)b * C_SZ + (size_t)g * CPG_SZ) * N_SZ;
  const float4* xg4 = (const float4*)xg;
  const int tid = threadIdx.x;

  float s = 0.0f, ss = 0.0f;
  for (int i = tid; i < (CPG_SZ * N_SZ) / 4; i += 256) {
    const float4 v = xg4[i];
    s  += v.x + v.y + v.z + v.w;
    ss += v.x * v.x + v.y * v.y + v.z * v.z + v.w * v.w;
  }
  __shared__ float r0[256], r1[256];
  r0[tid] = s; r1[tid] = ss;
  __syncthreads();
  for (int st = 128; st > 0; st >>= 1) {
    if (tid < st) { r0[tid] += r0[tid + st]; r1[tid] += r1[tid + st]; }
    __syncthreads();
  }
  const float inv_n = 1.0f / (float)(CPG_SZ * N_SZ);
  const float mean  = r0[0] * inv_n;
  const float var   = r1[0] * inv_n - mean * mean;
  const float rinv  = rsqrtf(var + 1e-5f);

  for (int i0 = tid * 4; i0 < CPG_SZ * N_SZ; i0 += 1024) {
    const float4 v4 = xg4[i0 >> 2];
    const int cc = i0 >> 10;
    const int c  = g * CPG_SZ + cc;
    const float sc = rinv * w[c];
    const float sb = bias[c] - mean * sc;
    const int nb = i0 & (N_SZ - 1);
    half_t* dst = xnT + ((size_t)b * N_SZ + nb) * C_SZ + c;
    dst[0 * C_SZ] = (half_t)(v4.x * sc + sb);
    dst[1 * C_SZ] = (half_t)(v4.y * sc + sb);
    dst[2 * C_SZ] = (half_t)(v4.z * sc + sb);
    dst[3 * C_SZ] = (half_t)(v4.w * sc + sb);
  }
}

// ---------------------------------------------------------------------------
// Kernel 2: QKV GEMMs. y[o,n] = sum_c W[o,c]*xn[c,n]+b[o]
// grid: x = N/128, y = C/64 (o-tile = head), z = B*3
// ---------------------------------------------------------------------------
__global__ void qkv_k(const half_t* __restrict__ xnT,
                      const float* __restrict__ qw, const float* __restrict__ qb,
                      const float* __restrict__ kw, const float* __restrict__ kb,
                      const float* __restrict__ vw, const float* __restrict__ vb,
                      half_t* __restrict__ qT, half_t* __restrict__ kT,
                      half_t* __restrict__ vW) {
  const int n0 = blockIdx.x * 128;
  const int o0 = blockIdx.y * 64;
  const int b  = blockIdx.z % B_SZ;
  const int which = blockIdx.z / B_SZ;
  const float* W  = (which == 0) ? qw : (which == 1) ? kw : vw;
  const float* Bi = (which == 0) ? qb : (which == 1) ? kb : vb;

  const int lane = threadIdx.x & 31, wave = threadIdx.x >> 5;
  const int wm = wave >> 1, wn = wave & 1;
  const float*  Arow  = W + (size_t)(o0 + wm * 16) * C_SZ;
  const half_t* Bbase = xnT + ((size_t)b * N_SZ + (size_t)(n0 + wn * 64)) * C_SZ;

  v8f acc[4] = { zero_v8f(), zero_v8f(), zero_v8f(), zero_v8f() };
  gemm_k_loop(Arow, Bbase, lane, acc);

  const int hh = lane >> 4, ncl = lane & 15;
  const int h = o0 / HD_SZ;
#pragma unroll
  for (int t = 0; t < 4; ++t) {
#pragma unroll
    for (int r = 0; r < 8; ++r) {
      const int o = o0 + wm * 16 + r + 8 * hh;
      const int n = n0 + wn * 64 + t * 16 + ncl;
      const float val = acc[t][r] + Bi[o];
      if (which == 2) {
        vW[((size_t)b * C_SZ + o) * N_SZ + n] = (half_t)val;
      } else {
        half_t* dst = (which == 0) ? qT : kT;
        const int ch = wm * 16 + r + 8 * hh;
        dst[(((size_t)b * NH_SZ + h) * N_SZ + n) * HD_SZ + ch] = (half_t)val;
      }
    }
  }
}

// ---------------------------------------------------------------------------
// Kernel 3: flash attention per (b,h). K/V tiles staged to LDS with async
// copies (double-buffered, ASYNCcnt); one wave owns 16 query rows; online
// softmax; P staged via LDS for the C->A layout change.
// grid: x = N/128, y = B*NH. Output attnT[b][n][c] (K-contig for proj GEMM).
// ---------------------------------------------------------------------------
__device__ __forceinline__ void stage_kv(const half_t* __restrict__ kbase,
                                         const half_t* __restrict__ vbase,
                                         int m0, half_t (*kt)[64], half_t (*vt)[64],
                                         int tid) {
#pragma unroll
  for (int u = 0; u < 2; ++u) {
    const int idx = tid + u * 256;      // 0..511 : 64 rows x 8 chunks-of-16B
    const int row = idx >> 3;
    const int co  = (idx & 7) * 8;      // halfs
    async_copy16(kbase + (size_t)(m0 + row) * HD_SZ + co, &kt[row][co]);
    async_copy16(vbase + (size_t)row * N_SZ + m0 + co,    &vt[row][co]);
  }
}

__global__ void attn_k(const half_t* __restrict__ qT, const half_t* __restrict__ kT,
                       const half_t* __restrict__ vW, half_t* __restrict__ aoT) {
  const int bh = blockIdx.y;
  const int b = bh / NH_SZ, h = bh % NH_SZ;
  const int tid = threadIdx.x;
  const int lane = tid & 31, wave = tid >> 5;
  const int nrow0 = blockIdx.x * 128 + wave * 16;

  __shared__ half_t ktile[2][64][64];   // [buf][m_local][c]   16 KB
  __shared__ half_t vtile[2][64][64];   // [buf][c_local][m]   16 KB
  __shared__ half_t pbuf[8][16][64];    // per-wave P staging  16 KB

  const half_t* qbase = qT + (((size_t)bh) * N_SZ + nrow0) * HD_SZ;
  const v16h qa0 = load_a_f16(qbase,      HD_SZ, lane);   // K = c 0..31
  const v16h qa1 = load_a_f16(qbase + 32, HD_SZ, lane);   // K = c 32..63
  const half_t* kbase = kT + ((size_t)bh) * N_SZ * HD_SZ;
  const half_t* vbase = vW + ((size_t)b * C_SZ + (size_t)h * HD_SZ) * N_SZ;

  float mrun[8], lrun[8];
#pragma unroll
  for (int r = 0; r < 8; ++r) { mrun[r] = -1e30f; lrun[r] = 0.0f; }
  v8f oacc[4] = { zero_v8f(), zero_v8f(), zero_v8f(), zero_v8f() };
  const float scale = 0.125f;  // HD^-0.5

  stage_kv(kbase, vbase, 0, ktile[0], vtile[0], tid);

  for (int it = 0; it < N_SZ / 64; ++it) {
    const int buf = it & 1;
    if (it + 1 < N_SZ / 64) {
      stage_kv(kbase, vbase, (it + 1) * 64, ktile[buf ^ 1], vtile[buf ^ 1], tid);
      WAIT_ASYNC(4);   // 4 ops just issued outstanding; current tile complete
    } else {
      WAIT_ASYNC(0);
    }
    __syncthreads();
    const half_t (*kt)[64] = ktile[buf];
    const half_t (*vt)[64] = vtile[buf];

    // S tile: 16 q-rows x 64 keys = 4 C-frags; reuse_a chains over qa0/qa1
    v8f s[4] = { zero_v8f(), zero_v8f(), zero_v8f(), zero_v8f() };
    s[0] = wmma_f16_ra(qa0, load_b_kc(&kt[0][0],  64, lane), s[0]);
    s[1] = wmma_f16_ra(qa0, load_b_kc(&kt[16][0], 64, lane), s[1]);
    s[2] = wmma_f16_ra(qa0, load_b_kc(&kt[32][0], 64, lane), s[2]);
    s[3] = wmma_f16   (qa0, load_b_kc(&kt[48][0], 64, lane), s[3]);
    s[0] = wmma_f16_ra(qa1, load_b_kc(&kt[0][0]  + 32, 64, lane), s[0]);
    s[1] = wmma_f16_ra(qa1, load_b_kc(&kt[16][0] + 32, 64, lane), s[1]);
    s[2] = wmma_f16_ra(qa1, load_b_kc(&kt[32][0] + 32, 64, lane), s[2]);
    s[3] = wmma_f16   (qa1, load_b_kc(&kt[48][0] + 32, 64, lane), s[3]);

    // scale + row max (reduce across the 16 lanes of this half-wave)
    float rowm[8];
#pragma unroll
    for (int r = 0; r < 8; ++r) {
#pragma unroll
      for (int t = 0; t < 4; ++t) s[t][r] *= scale;
      float mx = fmaxf(fmaxf(s[0][r], s[1][r]), fmaxf(s[2][r], s[3][r]));
#pragma unroll
      for (int off = 8; off >= 1; off >>= 1) mx = fmaxf(mx, __shfl_xor(mx, off, 32));
      rowm[r] = mx;
    }
    float alpha[8], rs[8];
#pragma unroll
    for (int r = 0; r < 8; ++r) {
      const float mnew = fmaxf(mrun[r], rowm[r]);
      alpha[r] = __expf(mrun[r] - mnew);
      mrun[r] = mnew;
      rs[r] = 0.0f;
    }
#pragma unroll
    for (int t = 0; t < 4; ++t)
#pragma unroll
      for (int r = 0; r < 8; ++r) {
        const float p = __expf(s[t][r] - mrun[r]);
        s[t][r] = p;
        rs[r] += p;
      }
#pragma unroll
    for (int r = 0; r < 8; ++r) {
#pragma unroll
      for (int off = 8; off >= 1; off >>= 1) rs[r] += __shfl_xor(rs[r], off, 32);
      lrun[r] = lrun[r] * alpha[r] + rs[r];
#pragma unroll
      for (int t = 0; t < 4; ++t) oacc[t][r] *= alpha[r];
    }
    // stage P (C-frag layout) to LDS, reload as A-frags (DS in-order per wave)
    {
      const int hh = lane >> 4, ncl = lane & 15;
#pragma unroll
      for (int t = 0; t < 4; ++t)
#pragma unroll
        for (int r = 0; r < 8; ++r)
          pbuf[wave][r + 8 * hh][t * 16 + ncl] = (half_t)s[t][r];
    }
    const v16h pa0 = load_a_f16(&pbuf[wave][0][0],      64, lane); // K = m 0..31
    const v16h pa1 = load_a_f16(&pbuf[wave][0][0] + 32, 64, lane); // K = m 32..63
    oacc[0] = wmma_f16_ra(pa0, load_b_kc(&vt[0][0],  64, lane), oacc[0]);
    oacc[1] = wmma_f16_ra(pa0, load_b_kc(&vt[16][0], 64, lane), oacc[1]);
    oacc[2] = wmma_f16_ra(pa0, load_b_kc(&vt[32][0], 64, lane), oacc[2]);
    oacc[3] = wmma_f16   (pa0, load_b_kc(&vt[48][0], 64, lane), oacc[3]);
    oacc[0] = wmma_f16_ra(pa1, load_b_kc(&vt[0][0]  + 32, 64, lane), oacc[0]);
    oacc[1] = wmma_f16_ra(pa1, load_b_kc(&vt[16][0] + 32, 64, lane), oacc[1]);
    oacc[2] = wmma_f16_ra(pa1, load_b_kc(&vt[32][0] + 32, 64, lane), oacc[2]);
    oacc[3] = wmma_f16   (pa1, load_b_kc(&vt[48][0] + 32, 64, lane), oacc[3]);
    __syncthreads();   // protect buf before it is overwritten next round
  }

  // finalize: divide by row sums, store attnT[b][n][c]
  const int hh = lane >> 4, ncl = lane & 15;
#pragma unroll
  for (int cs = 0; cs < 4; ++cs)
#pragma unroll
    for (int r = 0; r < 8; ++r) {
      const float val = oacc[cs][r] / lrun[r];
      const int n = nrow0 + r + 8 * hh;
      const int c = h * HD_SZ + cs * 16 + ncl;
      aoT[((size_t)b * N_SZ + n) * C_SZ + c] = (half_t)val;
    }
}

// ---------------------------------------------------------------------------
// Kernel 4: proj conv1x1 + bias + residual, f32 output.
// grid: x = N/128, y = C/64, z = B
// ---------------------------------------------------------------------------
__global__ void proj_k(const half_t* __restrict__ aoT, const float* __restrict__ pw,
                       const float* __restrict__ pb, const float* __restrict__ x,
                       float* __restrict__ out) {
  const int n0 = blockIdx.x * 128;
  const int o0 = blockIdx.y * 64;
  const int b  = blockIdx.z;
  const int lane = threadIdx.x & 31, wave = threadIdx.x >> 5;
  const int wm = wave >> 1, wn = wave & 1;
  const float*  Arow  = pw + (size_t)(o0 + wm * 16) * C_SZ;
  const half_t* Bbase = aoT + ((size_t)b * N_SZ + (size_t)(n0 + wn * 64)) * C_SZ;

  v8f acc[4] = { zero_v8f(), zero_v8f(), zero_v8f(), zero_v8f() };
  gemm_k_loop(Arow, Bbase, lane, acc);

  const int hh = lane >> 4, ncl = lane & 15;
#pragma unroll
  for (int t = 0; t < 4; ++t)
#pragma unroll
    for (int r = 0; r < 8; ++r) {
      const int o = o0 + wm * 16 + r + 8 * hh;
      const int n = n0 + wn * 64 + t * 16 + ncl;
      const size_t idx = ((size_t)b * C_SZ + o) * N_SZ + n;
      out[idx] = acc[t][r] + pb[o] + x[idx];
    }
}

// ---------------------------------------------------------------------------
extern "C" void kernel_launch(void* const* d_in, const int* in_sizes, int n_in,
                              void* d_out, int out_size, void* d_ws, size_t ws_size,
                              hipStream_t stream) {
  (void)in_sizes; (void)n_in; (void)out_size; (void)ws_size;
  const float* x  = (const float*)d_in[0];
  const float* nw = (const float*)d_in[1];
  const float* nb = (const float*)d_in[2];
  const float* qw = (const float*)d_in[3];
  const float* qb = (const float*)d_in[4];
  const float* kw = (const float*)d_in[5];
  const float* kb = (const float*)d_in[6];
  const float* vw = (const float*)d_in[7];
  const float* vb = (const float*)d_in[8];
  const float* pw = (const float*)d_in[9];
  const float* pb = (const float*)d_in[10];
  float* out = (float*)d_out;

  const size_t M = (size_t)B_SZ * N_SZ * C_SZ;  // 4.19M elems per f16 buffer
  half_t* xnT = (half_t*)d_ws;   // [B][N][C]
  half_t* qT  = xnT + M;         // [B][NH][N][HD]
  half_t* kT  = qT + M;          // [B][NH][N][HD]
  half_t* vWs = kT + M;          // [B][C][N]
  half_t* aoT = vWs + M;         // [B][N][C]

  groupnorm_k<<<dim3(B_SZ * G_SZ), 256, 0, stream>>>(x, nw, nb, xnT);
  qkv_k<<<dim3(N_SZ / 128, C_SZ / 64, B_SZ * 3), 256, 0, stream>>>(
      xnT, qw, qb, kw, kb, vw, vb, qT, kT, vWs);
  attn_k<<<dim3(N_SZ / 128, B_SZ * NH_SZ), 256, 0, stream>>>(qT, kT, vWs, aoT);
  proj_k<<<dim3(N_SZ / 128, C_SZ / 64, B_SZ), 256, 0, stream>>>(aoT, pw, pb, x, out);
}
